// HVAE_22720376996584
// MI455X (gfx1250) — matrix operand order repaired and checked by reading, not compiled, not run on hardware
//
#include <hip/hip_runtime.h>

typedef __attribute__((ext_vector_type(16))) __bf16 v16bf;
typedef __attribute__((ext_vector_type(8)))  __bf16 bf16x8;
typedef __attribute__((ext_vector_type(8)))  float  v8f;

#define USE_ASYNC 1

typedef __attribute__((address_space(3))) __bf16 lds_bf16;

// ---------------- helpers ----------------

__device__ __forceinline__ __bf16 f2b(float f) {
  // round-to-nearest-even f32 -> bf16
  unsigned u = __builtin_bit_cast(unsigned, f);
  unsigned r = (u + 0x7FFFu + ((u >> 16) & 1u)) >> 16;
  unsigned short s = (unsigned short)r;
  return __builtin_bit_cast(__bf16, s);
}

__device__ __forceinline__ float b2f(__bf16 b) {
  unsigned short s = __builtin_bit_cast(unsigned short, b);
  return __builtin_bit_cast(float, ((unsigned)s) << 16);
}

__device__ __forceinline__ float sigmoidf_(float x) {
  return 1.0f / (1.0f + __expf(-x));
}

__device__ __forceinline__ v8f wmma_bf16(v16bf a, v16bf b, v8f c) {
  return __builtin_amdgcn_wmma_f32_16x16x32_bf16(false, a, false, b,
                                                 (short)0, c, false, false);
}

// 16-byte global(bf16) -> LDS copy via CDNA5 async-to-LDS (ASYNCcnt-tracked).
__device__ __forceinline__ void cp16(__bf16* l, const __bf16* g) {
#if USE_ASYNC
  unsigned loff = (unsigned)(unsigned long long)(lds_bf16*)l;  // 32-bit LDS offset
  unsigned long long ga = (unsigned long long)g;               // global address
  asm volatile("global_load_async_to_lds_b128 %0, %1, off"
               :: "v"(loff), "v"(ga) : "memory");
#else
  *(bf16x8*)l = *(const bf16x8*)g;
#endif
}

// Counted ASYNCcnt wait: lit = allowed outstanding async ops (in-order completion
// means waiting to N leaves only the most recent N in flight).
#if USE_ASYNC
#define WAITA(lit) asm volatile("s_wait_asynccnt " lit ::: "memory")
#else
#define WAITA(lit)
#endif

// A fragment from LDS tile stored [16 rows][32 k] row-major.
// ISA 16-bit A 16x32: lane<16 -> K {0..7,16..23}; lane>=16 -> K {8..15,24..31}
__device__ __forceinline__ v16bf load_frag_a(const __bf16* t, int lane) {
  int r  = lane & 15;
  int kb = (lane >> 4) << 3;
  union { v16bf v; bf16x8 h[2]; } f;
  f.h[0] = *(const bf16x8*)(t + r * 32 + kb);
  f.h[1] = *(const bf16x8*)(t + r * 32 + kb + 16);
  return f.v;
}

// B fragment (32x16 KxN) from LDS tile stored [16 n][32 k].
// ISA: lanes 0-15 hold K {0..7,16..23} of column n, lanes 16-31 hold K {8..15,24..31}
__device__ __forceinline__ v16bf load_frag_b(const __bf16* t, int lane) {
  int n  = lane & 15;
  int kb = (lane >> 4) << 3;
  union { v16bf v; bf16x8 h[2]; } f;
  f.h[0] = *(const bf16x8*)(t + n * 32 + kb);
  f.h[1] = *(const bf16x8*)(t + n * 32 + kb + 16);
  return f.v;
}

// ---------------- one-shot weight conversion f32 -> bf16 ----------------

__global__ void conv_bf16(const float* __restrict__ s, __bf16* __restrict__ d, int n) {
  int i = blockIdx.x * 256 + threadIdx.x;
  if (i < n) d[i] = f2b(s[i]);
}

// ---------------- generic GEMM: C = A(rows,K)bf16 @ W(N,K)bf16^T + bias ----------------
// grid.x = rows/64, grid.y = N/32, block = 128 (4 waves). Double-buffered async tiles.

__global__ __launch_bounds__(128) void gemm_bias(
    const __bf16* __restrict__ A, int lda,
    const __bf16* __restrict__ W, const float* __restrict__ bias,
    float* __restrict__ C, int ldc, int K)
{
  __shared__ __align__(16) __bf16 AtB[2 * 2048];
  __shared__ __align__(16) __bf16 WtB[2 * 1024];
  const int t = threadIdx.x, wave = t >> 5, lane = t & 31;
  const int row0 = blockIdx.x * 64, n0 = blockIdx.y * 32;
  v8f z = {0, 0, 0, 0, 0, 0, 0, 0};
  v8f acc0 = z, acc1 = z;

  auto fill = [&](int buf, int k0) {
    __bf16* At = AtB + buf * 2048;
    __bf16* Wt = WtB + buf * 1024;
#pragma unroll
    for (int i = 0; i < 2; ++i) {
      int ch = i * 128 + t, r = ch >> 2, c = (ch & 3) * 8;
      cp16(At + r * 32 + c, A + (size_t)(row0 + r) * lda + k0 + c);
    }
    int r = t >> 2, c = (t & 3) * 8;
    cp16(Wt + r * 32 + c, W + (size_t)(n0 + r) * K + k0 + c);
  };

  fill(0, 0);
  for (int k0 = 0; k0 < K; k0 += 32) {
    int cur = (k0 >> 5) & 1;
    if (k0 + 32 < K) { fill(cur ^ 1, k0 + 32); WAITA("0x3"); }
    else             { WAITA("0x0"); }
    __syncthreads();
    const __bf16* At = AtB + cur * 2048;
    const __bf16* Wt = WtB + cur * 1024;
    v16bf a = load_frag_a(At + wave * 512, lane);
    acc0 = wmma_bf16(a, load_frag_b(Wt, lane), acc0);
    acc1 = wmma_bf16(a, load_frag_b(Wt + 512, lane), acc1);
    __syncthreads();
  }

  const int n = lane & 15, mb = (lane >> 4) << 3;
  const float b0 = bias[n0 + n], b1 = bias[n0 + 16 + n];
#pragma unroll
  for (int v = 0; v < 8; ++v) {
    int row = row0 + wave * 16 + mb + v;
    C[row * ldc + n0 + n]      = acc0[v] + b0;
    C[row * ldc + n0 + 16 + n] = acc1[v] + b1;
  }
}

// ---------------- encoder leaf level (d=6): children hidden = 0 ----------------

__global__ void enc_leaf(const int* __restrict__ symbols,
                         const float* __restrict__ ewi,
                         const float* __restrict__ ebi,
                         const float* __restrict__ ebh,
                         __bf16* __restrict__ h_out)
{
  int idx = blockIdx.x * 256 + threadIdx.x;        // 16384*512 threads
  int row = idx >> 9, j = idx & 511;
  int b = row >> 6, node = row & 63;
  int sym = symbols[b * 127 + 63 + node];
  float gir = ewi[j * 32 + sym] + ebi[j];
  float giz = ewi[(512 + j) * 32 + sym] + ebi[512 + j];
  float gin = ewi[(1024 + j) * 32 + sym] + ebi[1024 + j];
  float r  = sigmoidf_(gir + ebh[j]);
  float zg = sigmoidf_(giz + ebh[512 + j]);
  float nv = tanhf(gin + r * ebh[1024 + j]);
  h_out[row * 512 + j] = f2b((1.0f - zg) * nv);    // hl = hr = 0
}

// ---------------- fused encoder level: gh GEMM + GRU-merge epilogue ----------------
// h_in: (rows, 1024) bf16 = concat(child pairs); grid.x = rows/64, grid.y = 16

__global__ __launch_bounds__(128) void enc_level(
    const __bf16* __restrict__ h_in, const int* __restrict__ symbols,
    const float* __restrict__ ewi, const float* __restrict__ ebi,
    const __bf16* __restrict__ ewh_b, const float* __restrict__ ebh,
    __bf16* __restrict__ h_out, int n_d, int dshift)
{
  __shared__ __align__(16) __bf16 AtB[2 * 2048];
  __shared__ __align__(16) __bf16 WrB[2 * 1024];
  __shared__ __align__(16) __bf16 WzB[2 * 1024];
  __shared__ __align__(16) __bf16 WnB[2 * 1024];
  const int t = threadIdx.x, wave = t >> 5, lane = t & 31;
  const int row0 = blockIdx.x * 64, n0 = blockIdx.y * 32;
  v8f z = {0, 0, 0, 0, 0, 0, 0, 0};
  v8f ar[2] = {z, z}, az[2] = {z, z}, an[2] = {z, z};

  auto fill = [&](int buf, int k0) {
#pragma unroll
    for (int i = 0; i < 2; ++i) {
      int ch = i * 128 + t, r = ch >> 2, c = (ch & 3) * 8;
      cp16(AtB + buf * 2048 + r * 32 + c,
           h_in + (size_t)(row0 + r) * 1024 + k0 + c);
    }
    int r = t >> 2, c = (t & 3) * 8;
    cp16(WrB + buf * 1024 + r * 32 + c, ewh_b + (size_t)(n0 + r) * 1024 + k0 + c);
    cp16(WzB + buf * 1024 + r * 32 + c, ewh_b + (size_t)(512 + n0 + r) * 1024 + k0 + c);
    cp16(WnB + buf * 1024 + r * 32 + c, ewh_b + (size_t)(1024 + n0 + r) * 1024 + k0 + c);
  };

  fill(0, 0);
  for (int k0 = 0; k0 < 1024; k0 += 32) {
    int cur = (k0 >> 5) & 1;
    if (k0 + 32 < 1024) { fill(cur ^ 1, k0 + 32); WAITA("0x5"); }
    else                { WAITA("0x0"); }
    __syncthreads();
    v16bf a = load_frag_a(AtB + cur * 2048 + wave * 512, lane);
#pragma unroll
    for (int tl = 0; tl < 2; ++tl) {
      ar[tl] = wmma_bf16(a, load_frag_b(WrB + cur * 1024 + tl * 512, lane), ar[tl]);
      az[tl] = wmma_bf16(a, load_frag_b(WzB + cur * 1024 + tl * 512, lane), az[tl]);
      an[tl] = wmma_bf16(a, load_frag_b(WnB + cur * 1024 + tl * 512, lane), an[tl]);
    }
    __syncthreads();
  }

  const int nn = lane & 15, mb = (lane >> 4) << 3;
#pragma unroll
  for (int tl = 0; tl < 2; ++tl) {
    const int j = n0 + tl * 16 + nn;
    const float br = ebh[j], bz = ebh[512 + j], bn = ebh[1024 + j];
    const float cr = ebi[j], cz = ebi[512 + j], cn = ebi[1024 + j];
#pragma unroll
    for (int v = 0; v < 8; ++v) {
      int row = row0 + wave * 16 + mb + v;
      int b = row >> dshift, node = row & (n_d - 1);
      int sym = symbols[b * 127 + (n_d - 1) + node];
      float gir = ewi[j * 32 + sym] + cr;            // one-hot GEMM == gather
      float giz = ewi[(512 + j) * 32 + sym] + cz;
      float gin = ewi[(1024 + j) * 32 + sym] + cn;
      float r  = sigmoidf_(gir + ar[tl][v] + br);
      float zg = sigmoidf_(giz + az[tl][v] + bz);
      float nv = tanhf(gin + r * (an[tl][v] + bn));
      float hl = b2f(h_in[row * 1024 + j]);
      float hr = b2f(h_in[row * 1024 + 512 + j]);
      h_out[row * 512 + j] = f2b((1.0f - zg) * nv + zg * (hl + hr) * 0.5f);
    }
  }
}

// ---------------- reparameterization ----------------

__global__ void zlat_kernel(const float* __restrict__ mu,
                            const float* __restrict__ lv,
                            const float* __restrict__ eps,
                            __bf16* __restrict__ zlat)
{
  int i = blockIdx.x * 256 + threadIdx.x;            // 256*128 threads
  zlat[i] = f2b(mu[i] + eps[i] * __expf(0.5f * lv[i]));
}

// ---------------- pred scatter + softmax (bf16 A-matrix for dwi GEMM) ----------------

__global__ void pred_store(const float* __restrict__ predws,
                           float* __restrict__ preds,
                           __bf16* __restrict__ aws,
                           int n_d, int dshift, int do_softmax)
{
  int row = blockIdx.x * 256 + threadIdx.x;          // rows = 256<<d threads
  int b = row >> dshift, node = row & (n_d - 1);
  float vals[32];
  float mx = -1e30f;
  int base = row * 32;
#pragma unroll
  for (int s = 0; s < 32; ++s) { vals[s] = predws[base + s]; mx = fmaxf(mx, vals[s]); }
  int ob = b * (127 * 32) + ((n_d - 1) + node) * 32;
#pragma unroll
  for (int s = 0; s < 32; ++s) preds[ob + s] = vals[s];   // logits, pre-softmax
  if (do_softmax) {
    float sum = 0.0f;
#pragma unroll
    for (int s = 0; s < 32; ++s) { vals[s] = __expf(vals[s] - mx); sum += vals[s]; }
    float inv = 1.0f / sum;
#pragma unroll
    for (int s = 0; s < 32; ++s) aws[base + s] = f2b(vals[s] * inv);
  }
}

// ---------------- fused decoder level: dwh GEMM + dwi GEMM (K=32) + GRU split --------
// grid.x = rows/64, grid.y = 32 (covers j in [0,1024))

__global__ __launch_bounds__(128) void dec_level(
    const __bf16* __restrict__ hid,    // (rows, 512) bf16
    const __bf16* __restrict__ aws,    // (rows, 32) softmax(pred) bf16
    const __bf16* __restrict__ dwi_b,  // (3072, 32) bf16
    const float* __restrict__ dbi,
    const __bf16* __restrict__ dwh_b,  // (3072, 512) bf16
    const float* __restrict__ dbh,
    __bf16* __restrict__ h_next)       // (rows, 1024) == (2*rows, 512) children
{
  __shared__ __align__(16) __bf16 AtB[2 * 2048];
  __shared__ __align__(16) __bf16 WrB[2 * 1024];
  __shared__ __align__(16) __bf16 WzB[2 * 1024];
  __shared__ __align__(16) __bf16 WnB[2 * 1024];
  const int t = threadIdx.x, wave = t >> 5, lane = t & 31;
  const int row0 = blockIdx.x * 64, n0 = blockIdx.y * 32;
  v8f z = {0, 0, 0, 0, 0, 0, 0, 0};
  v8f ar[2] = {z, z}, az[2] = {z, z}, aghn[2] = {z, z}, agin[2] = {z, z};

  auto fill = [&](int buf, int k0) {
#pragma unroll
    for (int i = 0; i < 2; ++i) {
      int ch = i * 128 + t, r = ch >> 2, c = (ch & 3) * 8;
      cp16(AtB + buf * 2048 + r * 32 + c,
           hid + (size_t)(row0 + r) * 512 + k0 + c);
    }
    int r = t >> 2, c = (t & 3) * 8;
    cp16(WrB + buf * 1024 + r * 32 + c, dwh_b + (size_t)(n0 + r) * 512 + k0 + c);
    cp16(WzB + buf * 1024 + r * 32 + c, dwh_b + (size_t)(1024 + n0 + r) * 512 + k0 + c);
    cp16(WnB + buf * 1024 + r * 32 + c, dwh_b + (size_t)(2048 + n0 + r) * 512 + k0 + c);
  };

  // main K loop (pipelined): gh = hid @ dwh^T, strips j, j+2H, j+4H
  fill(0, 0);
  for (int k0 = 0; k0 < 512; k0 += 32) {
    int cur = (k0 >> 5) & 1;
    if (k0 + 32 < 512) { fill(cur ^ 1, k0 + 32); WAITA("0x5"); }
    else               { WAITA("0x0"); }
    __syncthreads();
    v16bf a = load_frag_a(AtB + cur * 2048 + wave * 512, lane);
#pragma unroll
    for (int tl = 0; tl < 2; ++tl) {
      ar[tl]   = wmma_bf16(a, load_frag_b(WrB + cur * 1024 + tl * 512, lane), ar[tl]);
      az[tl]   = wmma_bf16(a, load_frag_b(WzB + cur * 1024 + tl * 512, lane), az[tl]);
      aghn[tl] = wmma_bf16(a, load_frag_b(WnB + cur * 1024 + tl * 512, lane), aghn[tl]);
    }
    __syncthreads();
  }

  // one extra K-step: gi = a @ dwi^T (K=32). r/zg chunks fold into same
  // accumulators; n-chunk kept separate (needed as gi_n + r*gh_n).
  {
#pragma unroll
    for (int i = 0; i < 2; ++i) {
      int ch = i * 128 + t, r = ch >> 2, c = (ch & 3) * 8;
      cp16(AtB + r * 32 + c, aws + (size_t)(row0 + r) * 32 + c);
    }
    {
      int r = t >> 2, c = (t & 3) * 8;
      cp16(WrB + r * 32 + c, dwi_b + (size_t)(n0 + r) * 32 + c);
      cp16(WzB + r * 32 + c, dwi_b + (size_t)(1024 + n0 + r) * 32 + c);
      cp16(WnB + r * 32 + c, dwi_b + (size_t)(2048 + n0 + r) * 32 + c);
    }
    WAITA("0x0");
    __syncthreads();
    v16bf a = load_frag_a(AtB + wave * 512, lane);
#pragma unroll
    for (int tl = 0; tl < 2; ++tl) {
      ar[tl]   = wmma_bf16(a, load_frag_b(WrB + tl * 512, lane), ar[tl]);
      az[tl]   = wmma_bf16(a, load_frag_b(WzB + tl * 512, lane), az[tl]);
      agin[tl] = wmma_bf16(a, load_frag_b(WnB + tl * 512, lane), agin[tl]);
    }
    __syncthreads();
  }

  const int nn = lane & 15, mb = (lane >> 4) << 3;
#pragma unroll
  for (int tl = 0; tl < 2; ++tl) {
    const int j = n0 + tl * 16 + nn;
    const float br = dbi[j] + dbh[j];
    const float bz = dbi[1024 + j] + dbh[1024 + j];
    const float bhn = dbh[2048 + j], bin = dbi[2048 + j];
#pragma unroll
    for (int v = 0; v < 8; ++v) {
      int row = row0 + wave * 16 + mb + v;
      float r  = sigmoidf_(ar[tl][v] + br);
      float zg = sigmoidf_(az[tl][v] + bz);
      float nv = tanhf((agin[tl][v] + bin) + r * (aghn[tl][v] + bhn));
      float dh = b2f(hid[row * 512 + (j & 511)]);    // concat(hidden,hidden)
      h_next[row * 1024 + j] = f2b((1.0f - zg) * nv + zg * dh);
    }
  }
}

// ---------------- host orchestration ----------------

extern "C" void kernel_launch(void* const* d_in, const int* in_sizes, int n_in,
                              void* d_out, int out_size, void* d_ws, size_t ws_size,
                              hipStream_t stream) {
  const int*   symbols = (const int*)d_in[0];
  const float* eps     = (const float*)d_in[1];
  const float* ewi     = (const float*)d_in[2];
  const float* ebi     = (const float*)d_in[3];
  const float* ewh     = (const float*)d_in[4];
  const float* ebh     = (const float*)d_in[5];
  const float* h2mu_w  = (const float*)d_in[6];
  const float* h2mu_b  = (const float*)d_in[7];
  const float* h2lv_w  = (const float*)d_in[8];
  const float* h2lv_b  = (const float*)d_in[9];
  const float* z2h_w   = (const float*)d_in[10];
  const float* z2h_b   = (const float*)d_in[11];
  const float* h2o_w   = (const float*)d_in[12];
  const float* h2o_b   = (const float*)d_in[13];
  const float* dwi     = (const float*)d_in[14];
  const float* dbi     = (const float*)d_in[15];
  const float* dwh     = (const float*)d_in[16];
  const float* dbh     = (const float*)d_in[17];

  float* out = (float*)d_out;
  char* ws = (char*)d_ws;
  size_t o = 0;
  __bf16* hA      = (__bf16*)(ws + o); o += (size_t)16384 * 512 * 2;
  __bf16* hB      = (__bf16*)(ws + o); o += (size_t)16384 * 512 * 2;
  float*  predws  = (float*) (ws + o); o += (size_t)16384 * 32 * 4;
  __bf16* aws     = (__bf16*)(ws + o); o += (size_t)8192 * 32 * 2;
  __bf16* zlatb   = (__bf16*)(ws + o); o += (size_t)256 * 128 * 2;
  __bf16* ewh_b   = (__bf16*)(ws + o); o += (size_t)1536 * 1024 * 2;
  __bf16* dwh_b   = (__bf16*)(ws + o); o += (size_t)3072 * 512 * 2;
  __bf16* dwi_b   = (__bf16*)(ws + o); o += (size_t)3072 * 32 * 2;
  __bf16* h2mu_wb = (__bf16*)(ws + o); o += (size_t)128 * 512 * 2;
  __bf16* h2lv_wb = (__bf16*)(ws + o); o += (size_t)128 * 512 * 2;
  __bf16* z2h_wb  = (__bf16*)(ws + o); o += (size_t)512 * 128 * 2;
  __bf16* h2o_wb  = (__bf16*)(ws + o); o += (size_t)32 * 512 * 2;

  // ---- one-shot weight conversion to bf16 ----
  conv_bf16<<<(1536 * 1024) / 256, 256, 0, stream>>>(ewh,    ewh_b,   1536 * 1024);
  conv_bf16<<<(3072 * 512) / 256,  256, 0, stream>>>(dwh,    dwh_b,   3072 * 512);
  conv_bf16<<<(3072 * 32) / 256,   256, 0, stream>>>(dwi,    dwi_b,   3072 * 32);
  conv_bf16<<<(128 * 512) / 256,   256, 0, stream>>>(h2mu_w, h2mu_wb, 128 * 512);
  conv_bf16<<<(128 * 512) / 256,   256, 0, stream>>>(h2lv_w, h2lv_wb, 128 * 512);
  conv_bf16<<<(512 * 128) / 256,   256, 0, stream>>>(z2h_w,  z2h_wb,  512 * 128);
  conv_bf16<<<(32 * 512) / 256,    256, 0, stream>>>(h2o_w,  h2o_wb,  32 * 512);

  // ---- encoder: level 6 (children zero) then fused levels 5..0 ----
  enc_leaf<<<(16384 * 512) / 256, 256, 0, stream>>>(symbols, ewi, ebi, ebh, hA);
  __bf16* hin = hA; __bf16* hout = hB;
  for (int d = 5; d >= 0; --d) {
    int n_d = 1 << d, rows = 256 * n_d;
    enc_level<<<dim3(rows / 64, 16), 128, 0, stream>>>(hin, symbols, ewi, ebi,
                                                       ewh_b, ebh, hout, n_d, d);
    __bf16* tmp = hin; hin = hout; hout = tmp;
  }
  // root (256 x 512) is now in hin == hA

  // ---- latent heads: mu, logvar -> d_out; zlat; hidden0 -> hB ----
  gemm_bias<<<dim3(4, 4),  128, 0, stream>>>(hin, 512, h2mu_wb, h2mu_b, out,         128, 512);
  gemm_bias<<<dim3(4, 4),  128, 0, stream>>>(hin, 512, h2lv_wb, h2lv_b, out + 32768, 128, 512);
  zlat_kernel<<<(256 * 128) / 256, 256, 0, stream>>>(out, out + 32768, eps, zlatb);
  // hidden0 = zlat @ z2h^T + b: f32 into predws scratch, then convert to bf16 in hB
  gemm_bias<<<dim3(4, 16), 128, 0, stream>>>(zlatb, 128, z2h_wb, z2h_b,
                                             predws, 512, 128);
  conv_bf16<<<(256 * 512) / 256, 256, 0, stream>>>(predws, hB, 256 * 512);

  // ---- decoder: pred GEMM + scatter/softmax + fused GRU split per level ----
  __bf16* hid = hB; __bf16* hnx = hA;
  float* preds = out + 65536;
  for (int d = 0; d < 7; ++d) {
    int n_d = 1 << d, rows = 256 * n_d;
    gemm_bias<<<dim3(rows / 64, 1), 128, 0, stream>>>(hid, 512, h2o_wb, h2o_b,
                                                      predws, 32, 512);
    pred_store<<<rows / 256, 256, 0, stream>>>(predws, preds, aws, n_d, d,
                                               (d < 6) ? 1 : 0);
    if (d < 6) {
      dec_level<<<dim3(rows / 64, 32), 128, 0, stream>>>(hid, aws, dwi_b, dbi,
                                                         dwh_b, dbh, hnx);
      __bf16* tmp = hid; hid = hnx; hnx = tmp;
    }
  }
}
